// CP_Attention_Action_1563368096231
// MI455X (gfx1250) — matrix operand order repaired and checked by reading, not compiled, not run on hardware
//
#include <hip/hip_runtime.h>

typedef __attribute__((ext_vector_type(16))) _Float16 v16h;
typedef __attribute__((ext_vector_type(8)))  _Float16 v8h;
typedef __attribute__((ext_vector_type(8)))  float    v8f;

#define B_  8
#define N_  1024
#define DIM_ 512
#define H_  8
#define D_  64
#define QKVO 1536
#define SCALE_ 0.125f

#if defined(__HIP_DEVICE_COMPILE__) && __has_builtin(__builtin_amdgcn_global_load_async_to_lds_b128)
#define ASYNC_LDS 1
typedef int v4i_ __attribute__((ext_vector_type(4)));
typedef __attribute__((address_space(1))) v4i_* g_v4i_ptr;   // global (AS1) int4*
typedef __attribute__((address_space(3))) v4i_* l_v4i_ptr;   // LDS (AS3) int4*
__device__ __forceinline__ g_v4i_ptr to_global_v4i(const void* p) {
  return (g_v4i_ptr)(unsigned long long)(uintptr_t)p;
}
__device__ __forceinline__ l_v4i_ptr to_lds_v4i(const void* p) {
  // generic LDS pointer: low 32 bits are the LDS address (aperture rules)
  return (l_v4i_ptr)(unsigned int)(uintptr_t)p;
}
#endif

__device__ __forceinline__ void wait_async_lds() {
#ifdef ASYNC_LDS
#if __has_builtin(__builtin_amdgcn_s_wait_asynccnt)
  __builtin_amdgcn_s_wait_asynccnt(0);
#else
  asm volatile("s_wait_asynccnt 0" ::: "memory");
#endif
#endif
}

// ---------- WMMA fragment loaders (CDNA5 wave32 layouts, ISA 7.12.2) ----------

// A-matrix 16x32 f16 from row-major [ld], top-left at `base` (row0,k0 already applied).
// lane l: row = l&15 ; halves 0..7 = K[(l>>4)*8 .. +7], halves 8..15 = K[16+(l>>4)*8 .. +7]
__device__ __forceinline__ v16h load_frag_a(const _Float16* base, int ld, int lane) {
  int l2 = lane & 15, hi = lane >> 4;
  const _Float16* p = base + (size_t)l2 * ld + hi * 8;
  union { v16h v; v8h h[2]; } u;
  u.h[0] = *(const v8h*)(p);
  u.h[1] = *(const v8h*)(p + 16);
  return u.v;
}

// B-matrix 32x16 f16 where B[kk][n] = Wt[n][kk], Wt row-major [ld] (i.e. computing X @ Wt^T).
// lane l: col = l&15 ; halves 0..7 = K[(l>>4)*8 .. +7], halves 8..15 = K[16+(l>>4)*8 .. +7]
__device__ __forceinline__ v16h load_frag_b(const _Float16* base, int ld, int lane) {
  int l2 = lane & 15, hi = lane >> 4;
  const _Float16* p = base + (size_t)l2 * ld + hi * 8;
  union { v16h v; v8h h[2]; } u;
  u.h[0] = *(const v8h*)(p);
  u.h[1] = *(const v8h*)(p + 16);
  return u.v;
}

__device__ __forceinline__ float red_max16(float v) {
  v = fmaxf(v, __shfl_xor(v, 1, 32));
  v = fmaxf(v, __shfl_xor(v, 2, 32));
  v = fmaxf(v, __shfl_xor(v, 4, 32));
  v = fmaxf(v, __shfl_xor(v, 8, 32));
  return v;
}
__device__ __forceinline__ float red_sum16(float v) {
  v += __shfl_xor(v, 1, 32);
  v += __shfl_xor(v, 2, 32);
  v += __shfl_xor(v, 4, 32);
  v += __shfl_xor(v, 8, 32);
  return v;
}

// ---------- Precompute: fold CP branches into effective GEMM weights ----------

// CPc[f][r][s] = sum_t CP_C[r][s][t] * CP_att[t][f]
__global__ void k_cpc(const float* __restrict__ CP_C, const float* __restrict__ CP_att,
                      float* __restrict__ CPc) {
  int idx = blockIdx.x * 256 + threadIdx.x;
  if (idx >= 64 * 64 * 4) return;
  int f = idx & 3, s = (idx >> 2) & 63, r = idx >> 8;
  float acc = 0.f;
  for (int t = 0; t < 64; ++t) acc += CP_C[(r * 64 + s) * 64 + t] * CP_att[t * 4 + f];
  CPc[(f * 64 + r) * 64 + s] = acc;
}

// Mf[f][r][d] = sum_s CPc[f][r][s] * CP_V_w[d][s]
__global__ void k_mf(const float* __restrict__ CPc, const float* __restrict__ CP_V_w,
                     float* __restrict__ Mf) {
  int idx = blockIdx.x * 256 + threadIdx.x;
  if (idx >= 4 * 64 * 512) return;
  int d = idx & 511, r = (idx >> 9) & 63, f = idx >> 15;
  float acc = 0.f;
  for (int s = 0; s < 64; ++s) acc += CPc[(f * 64 + r) * 64 + s] * CP_V_w[d * 64 + s];
  Mf[(f * 64 + r) * 512 + d] = acc;
}

// W1h[o][c] = qkv_w[o][c] + sum_r CP_U_w[r][c] * Mf[o/512][r][o%512]   (f16)
// b1[o]    = CP_V_b[o%512] + sum_r CP_U_b[r] * Mf[o/512][r][o%512]
__global__ void k_w1(const float* __restrict__ qkv_w, const float* __restrict__ CP_U_w,
                     const float* __restrict__ CP_U_b, const float* __restrict__ CP_V_b,
                     const float* __restrict__ Mf, _Float16* __restrict__ W1h,
                     float* __restrict__ b1) {
  int idx = blockIdx.x * 256 + threadIdx.x;
  if (idx >= QKVO * 512) return;
  int c = idx & 511, o = idx >> 9;
  int f = o >> 9, d = o & 511;
  const float* M = Mf + (size_t)(f * 64) * 512 + d;  // stride 512 over r
  float acc = qkv_w[(size_t)o * 512 + c];
  for (int r = 0; r < 64; ++r) acc += CP_U_w[r * 512 + c] * M[r * 512];
  W1h[(size_t)o * 512 + c] = (_Float16)acc;
  if (c == 0) {
    float bb = CP_V_b[d];
    for (int r = 0; r < 64; ++r) bb += CP_U_b[r] * M[r * 512];
    b1[o] = bb;
  }
}

// W2h[o][c] = proj_w[o][c] + S * sum_r CP_U_w[r][c] * Mf[3][r][o]  (S == 1)
// b2[o]    = proj_b[o] + S * (CP_V_b[o] + sum_r CP_U_b[r] * Mf[3][r][o])
__global__ void k_w2(const float* __restrict__ proj_w, const float* __restrict__ proj_b,
                     const float* __restrict__ CP_U_w, const float* __restrict__ CP_U_b,
                     const float* __restrict__ CP_V_b, const float* __restrict__ Mf,
                     _Float16* __restrict__ W2h, float* __restrict__ b2) {
  int idx = blockIdx.x * 256 + threadIdx.x;
  if (idx >= 512 * 512) return;
  int c = idx & 511, o = idx >> 9;
  const float* M = Mf + (size_t)(3 * 64) * 512 + o;
  float acc = proj_w[(size_t)o * 512 + c];
  for (int r = 0; r < 64; ++r) acc += CP_U_w[r * 512 + c] * M[r * 512];
  W2h[(size_t)o * 512 + c] = (_Float16)acc;
  if (c == 0) {
    float bb = CP_V_b[o];
    for (int r = 0; r < 64; ++r) bb += CP_U_b[r] * M[r * 512];
    b2[o] = proj_b[o] + bb;
  }
}

__global__ void k_tohalf(const float* __restrict__ src, _Float16* __restrict__ dst, int n) {
  int i = blockIdx.x * 256 + threadIdx.x;
  if (i < n) dst[i] = (_Float16)src[i];
}

// ---------- WMMA GEMM: out(M,N) = A(M,K) @ W(N,K)^T + bias ----------
// One wave computes a 16x64 output tile; the 8 waves of a block share one
// 64-column weight tile (weight fragments stay hot in WGP$/L0).
// K-loop: manual ping-pong (2x unroll) -> loads fill the buffer NOT being
// consumed, so no register copies and WMMA overlaps in-flight loads.
__global__ __launch_bounds__(256) void gemm_wmma(const _Float16* __restrict__ A,
                                                 const _Float16* __restrict__ W,
                                                 const float* __restrict__ bias,
                                                 _Float16* __restrict__ outH,
                                                 float* __restrict__ outF,
                                                 int M, int N, int K, int storeHalf) {
  int lane = threadIdx.x & 31;
  int wib  = threadIdx.x >> 5;          // wave-in-block 0..7
  int colTiles = N >> 6;
  int ct   = blockIdx.x % colTiles;     // shared by all 8 waves of the block
  int rg   = blockIdx.x / colTiles;     // row group of 8 row tiles
  int r0   = (rg * 8 + wib) << 4;
  int c0   = ct << 6;
  int l2 = lane & 15, hi = lane >> 4;

  v8f acc[4] = {};
  const _Float16* Arow = A + (size_t)r0 * K;
  const _Float16* Wc   = W + (size_t)c0 * K;

  // prime stage A with k = 0
  v16h aA = load_frag_a(Arow, K, lane);
  v16h bA[4];
#pragma unroll
  for (int j = 0; j < 4; ++j) bA[j] = load_frag_b(Wc + (size_t)(j * 16) * K, K, lane);

  for (int k0 = 0; k0 < K; k0 += 64) {   // K is a multiple of 64
    int k1 = k0 + 32;
    if (k0 + 128 <= K) __builtin_prefetch(Wc + k0 + 64, 0, 1);
    // issue stage-B loads, then consume stage A
    v16h aB = load_frag_a(Arow + k1, K, lane);
    v16h bB[4];
#pragma unroll
    for (int j = 0; j < 4; ++j) bB[j] = load_frag_b(Wc + (size_t)(j * 16) * K + k1, K, lane);
#pragma unroll
    for (int j = 0; j < 4; ++j)
      acc[j] = __builtin_amdgcn_wmma_f32_16x16x32_f16(false, aA, false, bA[j],
                                                      (short)0, acc[j], false, false);
    // issue stage-A loads for next iteration, then consume stage B
    int k2 = (k0 + 64 < K) ? (k0 + 64) : 0;  // last-iter reload is harmless
    aA = load_frag_a(Arow + k2, K, lane);
#pragma unroll
    for (int j = 0; j < 4; ++j) bA[j] = load_frag_b(Wc + (size_t)(j * 16) * K + k2, K, lane);
#pragma unroll
    for (int j = 0; j < 4; ++j)
      acc[j] = __builtin_amdgcn_wmma_f32_16x16x32_f16(false, aB, false, bB[j],
                                                      (short)0, acc[j], false, false);
  }

#pragma unroll
  for (int j = 0; j < 4; ++j) {
    int col = c0 + j * 16 + l2;
    float bv = bias[col];
#pragma unroll
    for (int i = 0; i < 8; ++i) {
      int row = r0 + i + 8 * hi;  // C-layout: vgpr i -> row i + 8*(lane>=16)
      float v = acc[j][i] + bv;
      if (storeHalf) outH[(size_t)row * N + col] = (_Float16)v;
      else           outF[(size_t)row * N + col] = v;
    }
  }
}

// ---------- Fused flash attention (per b,h; WMMA scores + WMMA P·V) ----------
// Block: 128 threads = 4 waves, each wave owns 16 query rows (block covers 64).
__global__ __launch_bounds__(128) void attn_wmma(const _Float16* __restrict__ qkvh,
                                                 const unsigned char* __restrict__ mask,
                                                 _Float16* __restrict__ outh) {
  __shared__ _Float16 Kt[32][64];      // keys row-major [key][d]
  __shared__ _Float16 Vt[64][32];      // V transposed   [d][key]
  __shared__ _Float16 Ps[4][16][32];   // per-wave P tile (C->A layout round-trip)

  int tid = threadIdx.x;
  int lane = tid & 31, wave = tid >> 5;
  int l2 = lane & 15, hi = lane >> 4;

  int blk    = blockIdx.x;
  int rowblk = blk & 15;       // 1024/64 query chunks
  int bh     = blk >> 4;
  int b      = bh >> 3, h = bh & 7;
  int q0     = rowblk * 64 + wave * 16;

  const _Float16* base  = qkvh + (size_t)(b * N_) * QKVO + h * 64;
  const _Float16* qbase = base;
  const _Float16* kbase = base + 512;
  const _Float16* vbase = base + 1024;

  // q fragments for d=0..31 and d=32..63, reused for all key tiles
  v16h aq0 = load_frag_a(qbase + (size_t)q0 * QKVO,      QKVO, lane);
  v16h aq1 = load_frag_a(qbase + (size_t)q0 * QKVO + 32, QKVO, lane);

  v8f acc[4] = {};
  float mstat[8], lstat[8];
#pragma unroll
  for (int i = 0; i < 8; ++i) { mstat[i] = -1e30f; lstat[i] = 0.f; }

  for (int t = 0; t < N_ / 32; ++t) {
    int key0 = t * 32;
    __syncthreads();  // protect previous iteration's LDS reads
    {   // cooperative stage: K row-major (async DMA to LDS if available), V transposed
      int key = tid >> 2, dc = (tid & 3) * 16;
      const _Float16* ks = kbase + (size_t)(key0 + key) * QKVO + dc;
#ifdef ASYNC_LDS
      __builtin_amdgcn_global_load_async_to_lds_b128(
          to_global_v4i(ks),     to_lds_v4i(&Kt[key][dc]),     0, 0);
      __builtin_amdgcn_global_load_async_to_lds_b128(
          to_global_v4i(ks + 8), to_lds_v4i(&Kt[key][dc + 8]), 0, 0);
#else
      *(v8h*)&Kt[key][dc]     = *(const v8h*)ks;
      *(v8h*)&Kt[key][dc + 8] = *(const v8h*)(ks + 8);
#endif
      const _Float16* vs = vbase + (size_t)(key0 + key) * QKVO + dc;
      v8h v0 = *(const v8h*)vs;
      v8h v1 = *(const v8h*)(vs + 8);
#pragma unroll
      for (int j = 0; j < 8; ++j) Vt[dc + j][key]     = v0[j];
#pragma unroll
      for (int j = 0; j < 8; ++j) Vt[dc + 8 + j][key] = v1[j];
    }
    wait_async_lds();
    __syncthreads();

    // scores: S(16x32) = q(16x64) @ K_tile(32x64)^T, two 16-key subtiles
    v8f s0 = {}, s1 = {};
    {
      v16h b00 = load_frag_b(&Kt[0][0],   64, lane);
      v16h b01 = load_frag_b(&Kt[0][32],  64, lane);
      v16h b10 = load_frag_b(&Kt[16][0],  64, lane);
      v16h b11 = load_frag_b(&Kt[16][32], 64, lane);
      s0 = __builtin_amdgcn_wmma_f32_16x16x32_f16(false, aq0, false, b00, (short)0, s0, false, false);
      s0 = __builtin_amdgcn_wmma_f32_16x16x32_f16(false, aq1, false, b01, (short)0, s0, false, false);
      s1 = __builtin_amdgcn_wmma_f32_16x16x32_f16(false, aq0, false, b10, (short)0, s1, false, false);
      s1 = __builtin_amdgcn_wmma_f32_16x16x32_f16(false, aq1, false, b11, (short)0, s1, false, false);
    }
    float mb0 = mask[b * N_ + key0 + l2]      ? 0.f : -1e30f;
    float mb1 = mask[b * N_ + key0 + 16 + l2] ? 0.f : -1e30f;

    // online softmax update (row = i + 8*hi; reductions stay within 16-lane halves)
    float f0[8], f1[8];
#pragma unroll
    for (int i = 0; i < 8; ++i) {
      f0[i] = s0[i] * SCALE_ + mb0;
      f1[i] = s1[i] * SCALE_ + mb1;
      float mx = red_max16(fmaxf(f0[i], f1[i]));
      float mnew = fmaxf(mstat[i], mx);
      float alpha = __expf(mstat[i] - mnew);
      f0[i] = __expf(f0[i] - mnew);
      f1[i] = __expf(f1[i] - mnew);
      float rs = red_sum16(f0[i] + f1[i]);
      lstat[i] = lstat[i] * alpha + rs;
      mstat[i] = mnew;
#pragma unroll
      for (int j = 0; j < 4; ++j) acc[j][i] *= alpha;
    }

    // P: C-layout -> LDS -> A-layout (DS ops are in-order within a wave)
#pragma unroll
    for (int i = 0; i < 8; ++i) {
      Ps[wave][i + 8 * hi][l2]      = (_Float16)f0[i];
      Ps[wave][i + 8 * hi][16 + l2] = (_Float16)f1[i];
    }
    v16h aP = load_frag_a(&Ps[wave][0][0], 32, lane);

    // out(16x64) += P(16x32) @ V_tile(32x64): B frags read contiguously from Vt[d][key]
#pragma unroll
    for (int j = 0; j < 4; ++j) {
      v16h bv = load_frag_b(&Vt[j * 16][0], 32, lane);
      acc[j] = __builtin_amdgcn_wmma_f32_16x16x32_f16(false, aP, false, bv,
                                                      (short)0, acc[j], false, false);
    }
  }

  // normalize + store f16, layout [b][n][h*64+d] (GEMM-ready for proj)
#pragma unroll
  for (int j = 0; j < 4; ++j)
#pragma unroll
    for (int i = 0; i < 8; ++i) {
      int row = q0 + i + 8 * hi;
      int col = j * 16 + l2;
      float v = acc[j][i] / lstat[i];
      outh[(size_t)(b * N_ + row) * 512 + h * 64 + col] = (_Float16)v;
    }
}

// ---------- host launch ----------

extern "C" void kernel_launch(void* const* d_in, const int* in_sizes, int n_in,
                              void* d_out, int out_size, void* d_ws, size_t ws_size,
                              hipStream_t stream) {
  const float* x       = (const float*)d_in[0];
  const unsigned char* mask = (const unsigned char*)d_in[1];
  const float* qkv_w   = (const float*)d_in[2];
  const float* CP_U_w  = (const float*)d_in[3];
  const float* CP_U_b  = (const float*)d_in[4];
  const float* CP_V_w  = (const float*)d_in[5];
  const float* CP_V_b  = (const float*)d_in[6];
  const float* CP_C    = (const float*)d_in[7];
  const float* CP_att  = (const float*)d_in[8];
  const float* proj_w  = (const float*)d_in[9];
  const float* proj_b  = (const float*)d_in[10];

  char* ws = (char*)d_ws;
  float*    CPc  = (float*)(ws + 0);                 //  64 KB
  float*    Mf   = (float*)(ws + 65536);             // 512 KB
  float*    b1   = (float*)(ws + 589824);            //   6 KB
  float*    b2   = (float*)(ws + 595968);            //   2 KB
  _Float16* W1h  = (_Float16*)(ws + 598016);         // 1.5 MB
  _Float16* W2h  = (_Float16*)(ws + 2170880);        // 0.5 MB
  _Float16* xh   = (_Float16*)(ws + 2695168);        //   8 MB
  _Float16* qkvh = (_Float16*)(ws + 11083776);       //  24 MB
  _Float16* outh = (_Float16*)(ws + 36249600);       //   8 MB

  // 1. fold CP branches into effective weights
  k_cpc<<<64, 256, 0, stream>>>(CP_C, CP_att, CPc);
  k_mf <<<512, 256, 0, stream>>>(CPc, CP_V_w, Mf);
  k_w1 <<<3072, 256, 0, stream>>>(qkv_w, CP_U_w, CP_U_b, CP_V_b, Mf, W1h, b1);
  k_w2 <<<1024, 256, 0, stream>>>(proj_w, proj_b, CP_U_w, CP_U_b, CP_V_b, Mf, W2h, b2);
  k_tohalf<<<(B_ * N_ * DIM_ + 255) / 256, 256, 0, stream>>>(x, xh, B_ * N_ * DIM_);

  // 2. QKV GEMM: qkvh(8192x1536) = xh @ W1h^T + b1   (f16 out)
  {
    int M = B_ * N_, Nn = QKVO, K = DIM_;
    int blocks = (M / 16 / 8) * (Nn / 64);
    gemm_wmma<<<blocks, 256, 0, stream>>>(xh, W1h, b1, qkvh, nullptr, M, Nn, K, 1);
  }

  // 3. fused flash attention -> outh(8192x512) f16
  attn_wmma<<<B_ * H_ * (N_ / 64), 128, 0, stream>>>(qkvh, mask, outh);

  // 4. proj GEMM: d_out(8192x512) = outh @ W2h^T + b2   (f32 out)
  {
    int M = B_ * N_, Nn = DIM_, K = DIM_;
    int blocks = (M / 16 / 8) * (Nn / 64);
    gemm_wmma<<<blocks, 256, 0, stream>>>(outh, W2h, b2, nullptr, (float*)d_out, M, Nn, K, 0);
  }
}
